// RGCN_65266323030534
// MI455X (gfx1250) — compile-verified, matrix operand order checked by reading, not run on hardware
//
#include <hip/hip_runtime.h>

// ---------------------------------------------------------------------------
// RGCN (3 layers, 8 relations, F=128) for gfx1250 / MI455X.
// GEMMs via v_wmma_f32_16x16x32_bf16 with bf16 hi/lo split (fp32-class
// accuracy: hi*Hi + hi*Lo + lo*Hi, f32 accumulate).
// Weights staged in LDS per workgroup (bank-conflict-free padded layout) so
// B-operand traffic drops ~16x vs every wave re-reading W from L2.
// Edge aggregation via fp32 global atomics with precomputed 1/count weights.
// ---------------------------------------------------------------------------

typedef __attribute__((ext_vector_type(16))) __bf16 v16bf;
typedef __attribute__((ext_vector_type(8)))  float  v8f;

union FragBF16 {
    v16bf v;
    uint4 u[2];
};

// LDS layout constants: 128x128 bf16 matrix, row stride padded 128 -> 136
// halfs (272 B) so fragment loads are bank-conflict free across 64 banks.
#define LDS_ROW_STRIDE 136
#define LDS_HALF_SZ    17408      // 128 * 136 (halfs) per hi/lo plane
#define LDS_TOTAL_B    69632      // 2 * 17408 * sizeof(__bf16)

// ---------------- fp32 -> bf16 hi/lo split of activations ------------------
__global__ void split_kernel(const float* __restrict__ x,
                             __bf16* __restrict__ hi,
                             __bf16* __restrict__ lo, int n) {
    for (int i = blockIdx.x * blockDim.x + threadIdx.x; i < n;
         i += gridDim.x * blockDim.x) {
        float v = x[i];
        __bf16 h = (__bf16)v;
        hi[i] = h;
        lo[i] = (__bf16)(v - (float)h);
    }
}

// -------------- fused ReLU (in place) + bf16 hi/lo split -------------------
__global__ void relu_split_kernel(float* __restrict__ p,
                                  __bf16* __restrict__ hi,
                                  __bf16* __restrict__ lo, int n) {
    for (int i = blockIdx.x * blockDim.x + threadIdx.x; i < n;
         i += gridDim.x * blockDim.x) {
        float v = fmaxf(p[i], 0.0f);
        p[i] = v;
        __bf16 h = (__bf16)v;
        hi[i] = h;
        lo[i] = (__bf16)(v - (float)h);
    }
}

// --------- transpose 128x128 fp32 weight(s) into bf16 hi/lo slots ----------
// Input:  W[mat][k][n]  (row-major, k = Fin index, n = Fout index)
// Output: slot per matrix: hi[n*128+k] at base, lo[n*128+k] at +16384.
// Slot stride = 32768 bf16 elements (64 KB, densely packed for the LDS copy).
__global__ void wprep_kernel(const float* __restrict__ W,
                             __bf16* __restrict__ out, int nmat) {
    int total = nmat << 14;  // nmat * 128 * 128
    for (int i = blockIdx.x * blockDim.x + threadIdx.x; i < total;
         i += gridDim.x * blockDim.x) {
        int mat = i >> 14;
        int rem = i & 16383;
        int n = rem >> 7;
        int k = rem & 127;
        float v = W[((size_t)mat << 14) + (k << 7) + n];
        __bf16 h = (__bf16)v;
        __bf16* p = out + ((size_t)mat << 15) + (n << 7) + k;
        p[0]     = h;                       // hi
        p[16384] = (__bf16)(v - (float)h);  // lo
    }
}

// ---------------- per-(relation,dst) inverse edge counts -------------------
__global__ void zero_kernel(float* __restrict__ p, int n) {
    for (int i = blockIdx.x * blockDim.x + threadIdx.x; i < n;
         i += gridDim.x * blockDim.x)
        p[i] = 0.0f;
}

__global__ void count_kernel(const int* __restrict__ dst,
                             const int* __restrict__ etype,
                             float* __restrict__ cnt, int E, int N) {
    for (int e = blockIdx.x * blockDim.x + threadIdx.x; e < E;
         e += gridDim.x * blockDim.x)
        atomicAdd(&cnt[(size_t)etype[e] * N + dst[e]], 1.0f);
}

__global__ void recip_kernel(float* __restrict__ cnt, int n) {
    for (int i = blockIdx.x * blockDim.x + threadIdx.x; i < n;
         i += gridDim.x * blockDim.x)
        cnt[i] = 1.0f / fmaxf(cnt[i], 1.0f);
}

// ---------------------------- WMMA GEMM ------------------------------------
// C[N,128] = A[N,128] @ W[128,128] (+bias). A given as bf16 hi/lo in global;
// W given as a transposed hi/lo slot which is first copied into LDS with a
// padded, bank-conflict-free layout. Block = 512 threads = 16 waves covering
// 256 rows; each wave owns a 16-row strip x 128 cols (8 f32 accumulators).
// Per K-block of 32: 3 WMMAs per column tile (hi*Hi + hi*Lo + lo*Hi).
__global__ __launch_bounds__(512)
void gemm_kernel(const __bf16* __restrict__ Ahi, const __bf16* __restrict__ Alo,
                 const __bf16* __restrict__ Wslot, const float* __restrict__ bias,
                 float* __restrict__ C, int Nrows) {
    extern __shared__ __bf16 lds_w[];  // [2][128][LDS_ROW_STRIDE]

    // ---- cooperative copy: 64 KB weight slot -> padded LDS (all threads) --
    {
        const uint4* __restrict__ g = (const uint4*)Wslot;  // 4096 chunks
        const int tid = threadIdx.x;
#pragma unroll
        for (int i = 0; i < 8; ++i) {
            const int c = tid + i * 512;        // chunk id, 16 B each
            const int plane = c >> 11;          // 0 = hi, 1 = lo
            const int rem = c & 2047;
            const int row = rem >> 4;           // weight output-column index n
            const int cir = rem & 15;           // chunk within row
            uint4 t = g[c];
            *(uint4*)(lds_w + plane * LDS_HALF_SZ + row * LDS_ROW_STRIDE +
                      cir * 8) = t;
        }
    }
    __syncthreads();

    const int lane = threadIdx.x & 31;
    const int wave = threadIdx.x >> 5;
    const int m  = lane & 15;   // row within 16x16 fragment
    const int hp = lane >> 4;   // half-pair selector (K-chunk select)
    const int rowbase = blockIdx.x * 256 + wave * 16;
    if (rowbase >= Nrows) return;  // N is a multiple of 16; no barriers below

    const __bf16* Whi = lds_w;
    const __bf16* Wlo = lds_w + LDS_HALF_SZ;

    v8f acc[8];
#pragma unroll
    for (int t = 0; t < 8; ++t) acc[t] = (v8f){};

    const size_t arow = (size_t)(rowbase + m) * 128;

#pragma unroll
    for (int kb = 0; kb < 4; ++kb) {
        const int k0 = kb * 32 + hp * 8;  // first contiguous 8-half chunk
        FragBF16 ahi, alo;
        ahi.u[0] = *(const uint4*)(Ahi + arow + k0);
        ahi.u[1] = *(const uint4*)(Ahi + arow + k0 + 16);
        alo.u[0] = *(const uint4*)(Alo + arow + k0);
        alo.u[1] = *(const uint4*)(Alo + arow + k0 + 16);
#pragma unroll
        for (int t = 0; t < 8; ++t) {
            const int n = t * 16 + m;  // output column handled by this lane
            const int wb = n * LDS_ROW_STRIDE + k0;
            FragBF16 bhi, blo;
            bhi.u[0] = *(const uint4*)(Whi + wb);
            bhi.u[1] = *(const uint4*)(Whi + wb + 16);
            blo.u[0] = *(const uint4*)(Wlo + wb);
            blo.u[1] = *(const uint4*)(Wlo + wb + 16);
            acc[t] = __builtin_amdgcn_wmma_f32_16x16x32_bf16(
                false, ahi.v, false, bhi.v, (short)0, acc[t], false, false);
            acc[t] = __builtin_amdgcn_wmma_f32_16x16x32_bf16(
                false, ahi.v, false, blo.v, (short)0, acc[t], false, false);
            acc[t] = __builtin_amdgcn_wmma_f32_16x16x32_bf16(
                false, alo.v, false, bhi.v, (short)0, acc[t], false, false);
        }
    }

#pragma unroll
    for (int t = 0; t < 8; ++t) {
        const int col = t * 16 + m;
        const float b = bias ? bias[col] : 0.0f;
#pragma unroll
        for (int v = 0; v < 8; ++v) {
            const int r = rowbase + v + 8 * hp;  // C/D fragment row mapping
            C[(size_t)r * 128 + col] = acc[t][v] + b;
        }
    }
}

// --------------------- edge scatter (mean folded in) -----------------------
// One wave per edge (grid-stride). Lane holds a float4 (4 of 128 features).
__global__ void scatter_kernel(const float* __restrict__ H,
                               const int* __restrict__ eidx,
                               const int* __restrict__ etype,
                               const float* __restrict__ inv,
                               float* __restrict__ OUT,
                               int E, int N, int r) {
    const int lane = threadIdx.x & 31;
    const int wid = (blockIdx.x * blockDim.x + threadIdx.x) >> 5;
    const int nwaves = (gridDim.x * blockDim.x) >> 5;
    const int* __restrict__ src = eidx;
    const int* __restrict__ dst = eidx + E;
    for (int e = wid; e < E; e += nwaves) {
        if (etype[e] != r) continue;
        const int s = src[e];
        const int d = dst[e];
        const float w = inv[(size_t)r * N + d];
        const float4 h = *(const float4*)(H + (size_t)s * 128 + lane * 4);
        float* o = OUT + (size_t)d * 128 + lane * 4;
        atomicAdd(o + 0, h.x * w);
        atomicAdd(o + 1, h.y * w);
        atomicAdd(o + 2, h.z * w);
        atomicAdd(o + 3, h.w * w);
    }
}

__global__ void relu_kernel(float* __restrict__ p, int n) {
    for (int i = blockIdx.x * blockDim.x + threadIdx.x; i < n;
         i += gridDim.x * blockDim.x)
        p[i] = fmaxf(p[i], 0.0f);
}

// ---------------------------------------------------------------------------
extern "C" void kernel_launch(void* const* d_in, const int* in_sizes, int n_in,
                              void* d_out, int out_size, void* d_ws, size_t ws_size,
                              hipStream_t stream) {
    const float* x     = (const float*)d_in[0];
    const int*   eidx  = (const int*)d_in[1];   // [2,E]: src then dst
    const int*   etype = (const int*)d_in[2];
    const float* Wl[3]    = {(const float*)d_in[3], (const float*)d_in[6], (const float*)d_in[9]};
    const float* rootl[3] = {(const float*)d_in[4], (const float*)d_in[7], (const float*)d_in[10]};
    const float* bl[3]    = {(const float*)d_in[5], (const float*)d_in[8], (const float*)d_in[11]};

    const int N = in_sizes[0] / 128;
    const int E = in_sizes[2];

    // -------------------- workspace layout --------------------
    char* ws = (char*)d_ws;
    size_t off = 0;
    auto take = [&](size_t bytes) {
        void* p = ws + off;
        off = (off + bytes + 255) & ~(size_t)255;
        return p;
    };
    __bf16* wts  = (__bf16*)take((size_t)27 * 32768 * sizeof(__bf16)); // 27 weight slots
    float*  inv  = (float*)take((size_t)8 * N * sizeof(float));
    __bf16* Xhi  = (__bf16*)take((size_t)N * 128 * sizeof(__bf16));
    __bf16* Xlo  = (__bf16*)take((size_t)N * 128 * sizeof(__bf16));
    float*  Hbuf = (float*)take((size_t)N * 128 * sizeof(float));
    float*  Lbuf = (float*)take((size_t)N * 128 * sizeof(float));
    (void)ws_size; (void)n_in; (void)out_size;

    auto slot = [&](int layer, int mm) -> __bf16* {
        return wts + (size_t)(layer * 9 + mm) * 32768;  // mm==0: root, mm-1: relation
    };

    // -------------------- prep (cheap, rerun every call) -------------------
    for (int l = 0; l < 3; ++l) {
        wprep_kernel<<<128, 256, 0, stream>>>(rootl[l], slot(l, 0), 1);
        wprep_kernel<<<512, 256, 0, stream>>>(Wl[l],    slot(l, 1), 8);
    }
    zero_kernel<<<1024, 256, 0, stream>>>(inv, 8 * N);
    count_kernel<<<1024, 256, 0, stream>>>(eidx + E, etype, inv, E, N);
    recip_kernel<<<1024, 256, 0, stream>>>(inv, 8 * N);

    // -------------------- three RGCN layers --------------------
    const int gemm_grid = (N + 255) / 256;
    split_kernel<<<2048, 256, 0, stream>>>(x, Xhi, Xlo, N * 128);
    for (int l = 0; l < 3; ++l) {
        float* out = (l == 2) ? (float*)d_out : Lbuf;
        // root transform + bias (full overwrite -> initializes `out`)
        gemm_kernel<<<gemm_grid, 512, LDS_TOTAL_B, stream>>>(
            Xhi, Xlo, slot(l, 0), bl[l], out, N);
        // relation transforms + mean-scatter
        for (int r = 0; r < 8; ++r) {
            gemm_kernel<<<gemm_grid, 512, LDS_TOTAL_B, stream>>>(
                Xhi, Xlo, slot(l, 1 + r), nullptr, Hbuf, N);
            scatter_kernel<<<2048, 256, 0, stream>>>(Hbuf, eidx, etype, inv,
                                                     out, E, N, r);
        }
        if (l < 2) {
            // fused ReLU + next layer's bf16 hi/lo split (one pass)
            relu_split_kernel<<<2048, 256, 0, stream>>>(out, Xhi, Xlo, N * 128);
        } else {
            relu_kernel<<<2048, 256, 0, stream>>>(out, N * 128);
        }
    }
}
